// MultiHeadAttentionWithRope_15427522527509
// MI455X (gfx1250) — matrix operand order repaired and checked by reading, not compiled, not run on hardware
//
#include <hip/hip_runtime.h>

// ---------------------------------------------------------------------------
// MI455X (gfx1250, wave32) multi-head attention with RoPE.
// All three matmuls run on v_wmma_f32_16x16x32_bf16 (fp32 accumulate).
// Attention is flash-style: online softmax, scores never hit HBM.
// K tiles in attention are staged LDS-side by the Tensor Data Mover.
// ---------------------------------------------------------------------------

#define DI __device__ __forceinline__

typedef __attribute__((ext_vector_type(16))) __bf16        bf16x16;
typedef __attribute__((ext_vector_type(8)))  float         fx8;
typedef __attribute__((ext_vector_type(8)))  unsigned int  ux8;
typedef __attribute__((ext_vector_type(4)))  unsigned int  uix4;
typedef __attribute__((ext_vector_type(8)))  int           ix8;
typedef __attribute__((ext_vector_type(4)))  int           ix4;

constexpr int BB = 2, T = 2048, C = 2048, H = 16, D = 128;
constexpr int N3 = 3 * C;

// ---- bf16 helpers (round-to-nearest-even) ----------------------------------
DI unsigned short f2bf(float f) {
  union { float f; unsigned u; } v; v.f = f;
  unsigned r = v.u + 0x7FFFu + ((v.u >> 16) & 1u);
  return (unsigned short)(r >> 16);
}
DI float bf2f(unsigned short h) {
  union { unsigned u; float f; } v; v.u = ((unsigned)h) << 16; return v.f;
}
DI unsigned pack2(float lo, float hi) {
  return (unsigned)f2bf(lo) | ((unsigned)f2bf(hi) << 16);
}

DI fx8 wmma_bf16(ux8 a, ux8 b, fx8 c) {
  return __builtin_amdgcn_wmma_f32_16x16x32_bf16(
      false, __builtin_bit_cast(bf16x16, a),
      false, __builtin_bit_cast(bf16x16, b),
      (short)0, c, false, false);
}

// ---------------------------------------------------------------------------
// Kernel 0: fp32 -> bf16 conversion, 4 elements/thread
// ---------------------------------------------------------------------------
__global__ void cvt_f32_bf16_v4(const float* __restrict__ src,
                                unsigned short* __restrict__ dst, int n4) {
  int i = blockIdx.x * blockDim.x + threadIdx.x;
  if (i >= n4) return;
  float4 f = ((const float4*)src)[i];
  uint2 o; o.x = pack2(f.x, f.y); o.y = pack2(f.z, f.w);
  ((uint2*)dst)[i] = o;
}

// ---------------------------------------------------------------------------
// Kernel 1/3: bf16 GEMM  out[M x N] = A[M x K] * B[K x N],  K = 2048.
// Block = 256 threads (8 waves), tile 128(M) x 128(N) x 32(K).
// Waves laid out 4(M) x 2(N); each wave computes 32x64 via 2x4 WMMA tiles.
// Branch-free epilogue via template (bf16 vs fp32 output).
// ---------------------------------------------------------------------------
template <bool OUT_BF16>
__global__ void __launch_bounds__(256) gemm_bf16(
    const unsigned short* __restrict__ A,
    const unsigned short* __restrict__ Bm,
    int N, void* __restrict__ outp)
{
  constexpr int K = C;
  __shared__ unsigned short Alds[128][34];   // row-major, K contiguous (pad 34)
  __shared__ unsigned short Blds[128][34];   // transposed: [n][k], K contiguous

  const int tid = threadIdx.x, lane = tid & 31, wid = tid >> 5;
  const int hl = lane >> 4, ln = lane & 15;      // half-group / lane-in-group
  const int wm = wid >> 1, wn = wid & 1;
  const int mb = blockIdx.y * 128, nb = blockIdx.x * 128;

  const fx8 z = {0.f, 0.f, 0.f, 0.f, 0.f, 0.f, 0.f, 0.f};
  fx8 acc[2][4] = {{z, z, z, z}, {z, z, z, z}};

  for (int kb = 0; kb < K; kb += 32) {
    __syncthreads();
    // A tile: 128x32 halves = 2048 dwords, 8 per thread (coalesced b32)
#pragma unroll
    for (int it = 0; it < 8; ++it) {
      int u = tid + it * 256;
      int row = u >> 4, k2 = (u & 15) << 1;
      *(unsigned*)&Alds[row][k2] =
          *(const unsigned*)&A[(size_t)(mb + row) * K + kb + k2];
    }
    // B tile: 16x128 halves as dword loads, stored transposed (K contiguous)
#pragma unroll
    for (int it = 0; it < 4; ++it) {
      int u = tid + it * 256;
      int k = u >> 6, n = (u & 63) << 1;
      unsigned w = *(const unsigned*)&Bm[(size_t)(kb + k) * N + nb + n];
      Blds[n][k]     = (unsigned short)(w & 0xFFFF);
      Blds[n + 1][k] = (unsigned short)(w >> 16);
    }
    // prefetch next K tiles into cache (global_prefetch_b8)
    if (kb + 32 < K) {
      __builtin_prefetch(&A[(size_t)(mb + (tid >> 1)) * K + kb + 32 + (tid & 1) * 16], 0, 3);
      __builtin_prefetch(&Bm[(size_t)(kb + 32 + (tid >> 3)) * N + nb + (tid & 7) * 16], 0, 3);
    }
    __syncthreads();

    ux8 af[2], bfr[4];
#pragma unroll
    for (int i = 0; i < 2; ++i) {           // A fragments (ISA 16-bit A layout)
      int row = wm * 32 + i * 16 + ln;
#pragma unroll
      for (int v = 0; v < 8; ++v) {
        int kk = (v >> 2) * 16 + hl * 8 + (v & 3) * 2;
        af[i][v] = *(const unsigned*)&Alds[row][kk];
      }
    }
#pragma unroll
    for (int j = 0; j < 4; ++j) {           // B fragments (ISA B layout)
      int col = wn * 64 + j * 16 + ln;
#pragma unroll
      for (int v = 0; v < 8; ++v)
        bfr[j][v] = *(const unsigned*)&Blds[col][(v >> 2) * 16 + hl * 8 + (v & 3) * 2];
    }
#pragma unroll
    for (int i = 0; i < 2; ++i)
#pragma unroll
      for (int j = 0; j < 4; ++j)
        acc[i][j] = wmma_bf16(af[i], bfr[j], acc[i][j]);
  }

  // Epilogue: C/D layout -> global (lanes 0-15 rows 0-7, lanes 16-31 rows 8-15)
#pragma unroll
  for (int i = 0; i < 2; ++i)
#pragma unroll
    for (int j = 0; j < 4; ++j)
#pragma unroll
      for (int r = 0; r < 8; ++r) {
        int row = mb + wm * 32 + i * 16 + hl * 8 + r;
        int col = nb + wn * 64 + j * 16 + ln;
        size_t o = (size_t)row * N + col;
        if constexpr (OUT_BF16) ((unsigned short*)outp)[o] = f2bf(acc[i][j][r]);
        else                    ((float*)outp)[o]          = acc[i][j][r];
      }
}

// ---------------------------------------------------------------------------
// Kernel 2: RoPE on q,k (in-place on bf16 qkv buffer), 1/sqrt(D) folded into q
// ---------------------------------------------------------------------------
__global__ void rope_scale(unsigned short* __restrict__ qkv) {
  int idx = blockIdx.x * 256 + threadIdx.x;   // B*T*2*H*(D/2) = 8388608 exact
  int i  = idx & 63;
  int hh = (idx >> 6) & 15;
  int s  = (idx >> 10) & 1;
  int t  = (idx >> 11) & (T - 1);
  int b  = idx >> 22;
  size_t base = ((size_t)(b * T + t) * 3 + s) * C + hh * D + 2 * i;
  unsigned p = *(const unsigned*)&qkv[base];
  float x1 = bf2f((unsigned short)(p & 0xFFFF));
  float x2 = bf2f((unsigned short)(p >> 16));
  float inv = __expf(-(float)(2 * i) * (9.210340371976184f / 128.f)); // 10000^(-2i/D)
  float ang = (float)t * inv;
  float sn, cs; __sincosf(ang, &sn, &cs);
  float o1 = x1 * cs - x2 * sn;
  float o2 = x2 * cs + x1 * sn;
  if (s == 0) { o1 *= 0.08838834764831845f; o2 *= 0.08838834764831845f; } // 1/sqrt(128)
  *(unsigned*)&qkv[base] = pack2(o1, o2);
}

// ---------------------------------------------------------------------------
// Kernel 4: flash attention. Grid (T/128, H, B), 256 threads.
// Each wave owns a 16-row Q strip (Q kept in registers in A-fragment layout).
// K tiles staged to LDS by the Tensor Data Mover (pad_enable reproduces the
// [32][130] anti-bank-conflict padding); V staged transposed by VALU loads.
// ---------------------------------------------------------------------------
__global__ void __launch_bounds__(256) flash_attn(
    const unsigned short* __restrict__ qkv,  // [B*T][3][H][D] bf16 (roped)
    unsigned short* __restrict__ obuf)       // [B*T][H*D] bf16
{
  __shared__ unsigned short Klds[32][130];   // [kv][d], d contiguous (TDM pad)
  __shared__ unsigned short Vt[128][34];     // [d][kv], kv contiguous
  __shared__ unsigned short Plds[8][16][34]; // per-wave P staging [q][kv]

  const int tid = threadIdx.x, lane = tid & 31, wid = tid >> 5;
  const int hl = lane >> 4, ln = lane & 15;
  const int b = blockIdx.z, h = blockIdx.y, qt = blockIdx.x;
  const int qblk = qt * 128;
  const int qrow0 = qblk + wid * 16;
  const unsigned klds_base = (unsigned)(unsigned long long)(void*)&Klds[0][0];

  // Q fragments: 4 d-chunks of 16x32, loaded once, stay in registers
  ux8 qf[4];
  {
    size_t qb = ((size_t)(b * T + qrow0 + ln) * 3 + 0) * C + h * D;
#pragma unroll
    for (int c = 0; c < 4; ++c)
#pragma unroll
      for (int v = 0; v < 8; ++v) {
        int kk = c * 32 + (v >> 2) * 16 + hl * 8 + (v & 3) * 2;
        qf[c][v] = *(const unsigned*)&qkv[qb + kk];
      }
  }

  const fx8 z = {0.f, 0.f, 0.f, 0.f, 0.f, 0.f, 0.f, 0.f};
  fx8 acc[8];                    // 8 d-chunks of 16x16 fp32 output
#pragma unroll
  for (int c = 0; c < 8; ++c) acc[c] = z;
  float m[8], l[8];
#pragma unroll
  for (int r = 0; r < 8; ++r) { m[r] = -__builtin_inff(); l[r] = 0.f; }

  const int kmax = qblk + 128;   // causal upper bound (block-uniform)
  for (int kb = 0; kb < kmax; kb += 32) {
    __syncthreads();
    // ---- K tile 32x128 via Tensor Data Mover (wave 0 issues; EXEC ignored).
    // D# group0: count=1 | lds_addr | 57-bit global_addr | type=2.
    // D# group1: data_size=2B, pad_enable, pad_interval=64dw, pad_amount=1dw
    //            (-> LDS row stride 130 halves), dims 128x32, stride 3*C.
    if (wid == 0) {
      unsigned long long ga = (unsigned long long)(const void*)
          &qkv[((size_t)(b * T + kb) * 3 + 1) * C + h * D];
      uix4 g0;
      g0[0] = 1u;                                   // count=1, user mode
      g0[1] = klds_base;                            // lds_addr (bytes)
      g0[2] = (unsigned)(ga & 0xFFFFFFFFu);         // global_addr[31:0]
      g0[3] = (unsigned)((ga >> 32) & 0x1FFFFFFu)   // global_addr[56:32]
            | (2u << 30);                           // type = 2 ("image")
      ix8 g1;
      g1[0] = (int)((1u << 16)      // data_size = 2 bytes
                  | (1u << 20)      // pad_enable
                  | (5u << 22));    // pad_interval: 64 dwords; pad_amount: 1 dword
      g1[1] = (int)(128u << 16);    // tensor_dim0 = 128 (d)
      g1[2] = (int)(32u  << 16);    // tensor_dim1 = 32  (kv)
      g1[3] = (int)(128u << 16);    // tile_dim0 = 128
      g1[4] = 32;                   // tile_dim1 = 32, tile_dim2 = 0
      g1[5] = (int)(3u * C);        // tensor_dim0_stride = 6144 elems
      g1[6] = 0; g1[7] = 0;
      ix4 gz4 = {0, 0, 0, 0};
      ix8 gz8 = {0, 0, 0, 0, 0, 0, 0, 0};
      __builtin_amdgcn_tensor_load_to_lds(g0, g1, gz4, gz4, gz8, 0);
      __builtin_amdgcn_s_wait_tensorcnt(0);
    }
    // ---- V tile 32x128 stored transposed [d][kv] (all threads)
#pragma unroll
    for (int it = 0; it < 16; ++it) {
      int u = tid + it * 256;
      int kv = u >> 7, d = u & 127;
      size_t g = ((size_t)(b * T + kb + kv) * 3 + 2) * C + h * D + d;
      Vt[d][kv] = qkv[g];
    }
    __syncthreads();

    // ---- scores: S = Q K^T (two 16x16 tiles, K-dim = D via 4 WMMAs each)
    fx8 s0 = z, s1 = z;
#pragma unroll
    for (int c = 0; c < 4; ++c) {
      ux8 b0, b1;
#pragma unroll
      for (int v = 0; v < 8; ++v) {
        int kk = c * 32 + (v >> 2) * 16 + hl * 8 + (v & 3) * 2;
        b0[v] = *(const unsigned*)&Klds[ln][kk];        // cols kb..kb+15
        b1[v] = *(const unsigned*)&Klds[16 + ln][kk];   // cols kb+16..kb+31
      }
      s0 = wmma_bf16(qf[c], b0, s0);
      s1 = wmma_bf16(qf[c], b1, s1);
    }

    // ---- causal mask + online softmax (row stats via shfl butterflies)
#pragma unroll
    for (int r = 0; r < 8; ++r) {
      int qr = qrow0 + hl * 8 + r;
      float v0 = (kb + ln      <= qr) ? s0[r] : -__builtin_inff();
      float v1 = (kb + 16 + ln <= qr) ? s1[r] : -__builtin_inff();
      float mx = fmaxf(v0, v1);
      mx = fmaxf(mx, __shfl_xor(mx, 1, 32));
      mx = fmaxf(mx, __shfl_xor(mx, 2, 32));
      mx = fmaxf(mx, __shfl_xor(mx, 4, 32));
      mx = fmaxf(mx, __shfl_xor(mx, 8, 32));
      float mn = fmaxf(m[r], mx);
      float alpha = __expf(m[r] - mn);
      float p0 = __expf(v0 - mn);
      float p1 = __expf(v1 - mn);
      float rs = p0 + p1;
      rs += __shfl_xor(rs, 1, 32);
      rs += __shfl_xor(rs, 2, 32);
      rs += __shfl_xor(rs, 4, 32);
      rs += __shfl_xor(rs, 8, 32);
      l[r] = l[r] * alpha + rs;
      m[r] = mn;
#pragma unroll
      for (int c = 0; c < 8; ++c) acc[c][r] *= alpha;
      Plds[wid][hl * 8 + r][ln]      = f2bf(p0);
      Plds[wid][hl * 8 + r][16 + ln] = f2bf(p1);
    }
    // same-wave cross-lane LDS RAW: drain DScnt before re-reading P
    asm volatile("s_wait_dscnt 0x0" ::: "memory");

    // ---- O += P V  (A = P 16x32, B = V 32x16 per d-chunk)
    ux8 pf;
#pragma unroll
    for (int v = 0; v < 8; ++v) {
      int kk = (v >> 2) * 16 + hl * 8 + (v & 3) * 2;
      pf[v] = *(const unsigned*)&Plds[wid][ln][kk];
    }
#pragma unroll
    for (int c = 0; c < 8; ++c) {
      ux8 vf;
#pragma unroll
      for (int v = 0; v < 8; ++v)
        vf[v] = *(const unsigned*)&Vt[c * 16 + ln][(v >> 2) * 16 + hl * 8 + (v & 3) * 2];
      acc[c] = wmma_bf16(pf, vf, acc[c]);
    }
  }

  // ---- normalize + store O (bf16)
#pragma unroll
  for (int r = 0; r < 8; ++r) {
    float inv_l = 1.0f / l[r];
    int qr = qrow0 + hl * 8 + r;
    size_t base = (size_t)(b * T + qr) * C + h * D;
#pragma unroll
    for (int c = 0; c < 8; ++c)
      obuf[base + c * 16 + ln] = f2bf(acc[c][r] * inv_l);
  }
}

// ---------------------------------------------------------------------------
// Launch: cvt x3 -> gemm(qkv) -> rope -> flash_attn -> gemm(out, fp32)
// Workspace map (bytes):
//   [0,16M) xb | [16M,41.9M) wqkvb | [41.9M,50.3M) woutb
//   [50.3M,100.7M) qkv | [100.7M,117.4M) obuf        total 112 MiB
// ---------------------------------------------------------------------------
extern "C" void kernel_launch(void* const* d_in, const int* in_sizes, int n_in,
                              void* d_out, int out_size, void* d_ws, size_t ws_size,
                              hipStream_t stream) {
  const float* x    = (const float*)d_in[0];
  const float* Wqkv = (const float*)d_in[1];
  const float* Wout = (const float*)d_in[2];
  char* ws = (char*)d_ws;
  unsigned short* xb    = (unsigned short*)(ws);
  unsigned short* wqkvb = (unsigned short*)(ws + (size_t)16777216);
  unsigned short* woutb = (unsigned short*)(ws + (size_t)41943040);
  unsigned short* qkvb  = (unsigned short*)(ws + (size_t)50331648);
  unsigned short* obuf  = (unsigned short*)(ws + (size_t)100663296);

  cvt_f32_bf16_v4<<<(BB * T * C / 4 + 255) / 256, 256, 0, stream>>>(x, xb, BB * T * C / 4);
  cvt_f32_bf16_v4<<<(C * N3 / 4 + 255) / 256, 256, 0, stream>>>(Wqkv, wqkvb, C * N3 / 4);
  cvt_f32_bf16_v4<<<(C * C / 4 + 255) / 256, 256, 0, stream>>>(Wout, woutb, C * C / 4);

  gemm_bf16<true><<<dim3(N3 / 128, BB * T / 128), 256, 0, stream>>>(
      xb, wqkvb, N3, qkvb);

  rope_scale<<<(BB * T * 2 * H * (D / 2)) / 256, 256, 0, stream>>>(qkvb);

  flash_attn<<<dim3(T / 128, H, BB), 256, 0, stream>>>(qkvb, obuf);

  gemm_bf16<false><<<dim3(C / 128, BB * T / 128), 256, 0, stream>>>(
      obuf, woutb, C, (float*)d_out);
}
